// Model_35931696398573
// MI455X (gfx1250) — matrix operand order repaired and checked by reading, not compiled, not run on hardware
//
#include <hip/hip_runtime.h>
#include <hip/hip_bf16.h>

// ---------------------------------------------------------------------------
// CDNA5 (gfx1250) DGCNN forward, wave32 + WMMA f16 16x16x32.
// Fused kNN (distance tiles never hit memory), WMMA edge-convs with
// max-over-k in registers, WMMA pointwise-MLP chain staged through LDS
// with GLOBAL_LOAD_ASYNC_TO_LDS_B128 (ASYNCcnt) when available.
// ---------------------------------------------------------------------------

typedef __attribute__((ext_vector_type(16))) _Float16 v16h;
typedef __attribute__((ext_vector_type(8)))  float    v8f;
typedef __attribute__((ext_vector_type(2)))  _Float16 h2;
typedef __attribute__((ext_vector_type(4)))  int      i4;

#define KNN 10
#define BATCH 8
// scale = g / sqrt(1 + 1e-5)
#define BN_RSQRT 0.9999950000374997f

#if __has_builtin(__builtin_amdgcn_global_load_async_to_lds_b128) && \
    __has_builtin(__builtin_amdgcn_s_wait_asynccnt)
#define HAVE_ASYNC_LDS 1
typedef __attribute__((address_space(1))) i4* gi4_ptr;   // global int4*
typedef __attribute__((address_space(3))) i4* li4_ptr;   // LDS int4*
#else
#define HAVE_ASYNC_LDS 0
#endif

// mish(x) = x * tanh(softplus(x)) = x * (e^2 + 2e) / (e^2 + 2e + 2), e = exp(x)
// Single transcendental + v_rcp; e clamped so e*e stays finite (saturates to x).
__device__ __forceinline__ float mish_f(float x) {
    float e = __expf(x);
    e = fminf(e, 1.0e18f);
    float t = e * (e + 2.0f);
    return x * t * __builtin_amdgcn_rcpf(t + 2.0f);
}

// ---- WMMA fragment loaders (contiguous k = 16*(l>>4) + s in A and B, wave32) ---------

// A-matrix 16x32 (MxK), source row-major [16][>=32] with given element stride.
// lane l: M = l&15; group g = l>>4; VGPR v holds K pair:
//   k0 = 2v + 16*g
__device__ __forceinline__ v16h load_frag_a(const _Float16* base, int stride, int lane) {
    v16h f;
    const _Float16* p = base + (size_t)(lane & 15) * stride;
    int g = (lane >> 4) * 16;
#pragma unroll
    for (int v = 0; v < 8; ++v) {
        int k0 = 2 * v + g;
        h2 pr = *(const h2*)(p + k0);
        f[2 * v]     = pr[0];
        f[2 * v + 1] = pr[1];
    }
    return f;
}

// B-matrix 32x16 (KxN), source stored "point-major": per column n a contiguous
// feature vector with given stride. lane l: N = l&15; k0 = 2v + 16*(l>>4).
__device__ __forceinline__ v16h load_frag_b(const _Float16* base, int stride, int lane) {
    v16h f;
    const _Float16* p = base + (size_t)(lane & 15) * stride;
    int g = (lane >> 4) * 16;
#pragma unroll
    for (int v = 0; v < 8; ++v) {
        int k0 = 2 * v + g;
        h2 pr = *(const h2*)(p + k0);
        f[2 * v]     = pr[0];
        f[2 * v + 1] = pr[1];
    }
    return f;
}

__device__ __forceinline__ v8f wmma_f16(v16h a, v16h b, v8f c) {
    return __builtin_amdgcn_wmma_f32_16x16x32_f16(
        /*neg_a=*/false, a, /*neg_b=*/false, b,
        /*c_mod=*/(short)0, c, /*reuse_a=*/false, /*reuse_b=*/false);
}

// Edge-feature B fragment: edge = [nbr - ctr (CP ch, zero-padded) ; ctr (CP ch)]
template<int CP>
__device__ __forceinline__ v16h make_edge_bfrag(const _Float16* ctr, const _Float16* nbr,
                                                int kc, int lane) {
    v16h f;
    int g = (lane >> 4) * 16;
    if (kc < CP) {
#pragma unroll
        for (int v = 0; v < 8; ++v) {
            int k0 = kc + 2 * v + g;
            h2 a = *(const h2*)(nbr + k0);
            h2 c = *(const h2*)(ctr + k0);
            f[2 * v]     = (_Float16)(a[0] - c[0]);
            f[2 * v + 1] = (_Float16)(a[1] - c[1]);
        }
    } else {
#pragma unroll
        for (int v = 0; v < 8; ++v) {
            int k0 = kc - CP + 2 * v + g;
            h2 c = *(const h2*)(ctr + k0);
            f[2 * v]     = c[0];
            f[2 * v + 1] = c[1];
        }
    }
    return f;
}

// Stage a [points][CH] f16 tile from global into LDS (row stride dstStride),
// using gfx1250 async memory->LDS B128 copies when the builtin exists.
// Pointer->integer->pointer round-trips give the exact address-space pointers
// the builtin wants (flat-to-LDS: low 32 bits are the LDS byte offset).
template<int CH>
__device__ __forceinline__ void stage_to_lds(_Float16* dst, const _Float16* src,
                                             int points, int dstStride, int lane) {
#if HAVE_ASYNC_LDS
    const int chunksPerRow = CH / 8;              // 8 halves = 16 B per async op
    const int chunks = points * chunksPerRow;
    for (int t = lane; t < chunks; t += 32) {
        int p = t / chunksPerRow;
        int c = (t - p * chunksPerRow) * 8;
        const _Float16* s = src + (size_t)p * CH + c;
        _Float16*       d = dst + (size_t)p * dstStride + c;
        __builtin_amdgcn_global_load_async_to_lds_b128(
            (gi4_ptr)(unsigned long long)(size_t)s,
            (li4_ptr)(unsigned)(size_t)d,
            0, 0);
    }
    __builtin_amdgcn_s_wait_asynccnt(0);
#else
    for (int t = lane; t < points * CH; t += 32) {
        int p = t / CH, c = t - p * CH;
        dst[(size_t)p * dstStride + c] = src[(size_t)p * CH + c];
    }
#endif
}

// ---------------------------------------------------------------------------
// Kernel 0: pack weights f32 (OxI) -> f16 padded [O][IP], zero-filled.
// split>0: input is [diff(split) ; ctr(split)] remapped to [0..) and [CP..).
// ---------------------------------------------------------------------------
__global__ void prep_weight_kernel(const float* __restrict__ W, _Float16* __restrict__ Wp,
                                   int O, int I, int IP, int split, int CP) {
    int t = blockIdx.x * blockDim.x + threadIdx.x;
    if (t >= O * IP) return;
    int o = t / IP, ip = t - o * IP;
    float v = 0.0f;
    if (split > 0) {
        if (ip < CP) { if (ip < split) v = W[o * I + ip]; }
        else         { int r = ip - CP; if (r < split) v = W[o * I + split + r]; }
    } else {
        if (ip < I) v = W[o * I + ip];
    }
    Wp[t] = (_Float16)v;
}

// ---------------------------------------------------------------------------
// Kernel 1: x (B,3,N) f32 -> xh (B,N,32) f16 zero-padded + xx (B,N) f32
// ---------------------------------------------------------------------------
__global__ void convert_x_kernel(const float* __restrict__ x, _Float16* __restrict__ xh,
                                 float* __restrict__ xx, int N) {
    int t = blockIdx.x * blockDim.x + threadIdx.x;   // over B*N
    int b = t / N, n = t - b * N;
    const float* xb = x + (size_t)b * 3 * N + n;
    float s = 0.0f;
#pragma unroll
    for (int c = 0; c < 32; ++c) {
        float v = (c < 3) ? xb[(size_t)c * N] : 0.0f;
        xh[(size_t)t * 32 + c] = (_Float16)v;
        if (c < 3) s += v * v;
    }
    xx[t] = s;
}

// ---------------------------------------------------------------------------
// Kernel 2: fused kNN. Each wave owns 16 query points; streams 16-candidate
// tiles through WMMA; per-lane sorted top-10 in registers; lane/lane+16 lists
// merged via LDS. Distance matrix never touches memory.
// Block = 64 threads (2 waves, each its own tile).
// ---------------------------------------------------------------------------
template<int CP>
__global__ void knn_kernel(const _Float16* __restrict__ xh, const float* __restrict__ xx,
                           int* __restrict__ idx, int N) {
    int wave = threadIdx.x >> 5, lane = threadIdx.x & 31;
    int tile = blockIdx.x * 2 + wave;
    int tilesPerB = N / 16;
    int b = tile / tilesPerB, qt = tile - b * tilesPerB;

    const _Float16* xb  = xh + (size_t)b * N * CP;
    const float*    xxb = xx + (size_t)b * N;
    int q = qt * 16 + (lane & 15);
    float xxq = xxb[q];

    // Query B-fragments are loop-invariant: hoist all K chunks.
    v16h bq[CP / 32];
#pragma unroll
    for (int kc = 0; kc < CP / 32; ++kc)
        bq[kc] = load_frag_b(xb + (size_t)qt * 16 * CP + kc * 32, CP, lane);

    float bestd[KNN];
    int   besti[KNN];
#pragma unroll
    for (int s = 0; s < KNN; ++s) { bestd[s] = -3.0e38f; besti[s] = 0; }

    for (int ct = 0; ct < N / 16; ++ct) {
        v8f acc = {};
#pragma unroll
        for (int kc = 0; kc < CP / 32; ++kc) {
            v16h a = load_frag_a(xb + (size_t)ct * 16 * CP + kc * 32, CP, lane);
            acc = wmma_f16(a, bq[kc], acc);
        }
#pragma unroll
        for (int r = 0; r < 8; ++r) {
            int m = ct * 16 + ((lane >> 4) << 3) + r;
            float negd = 2.0f * acc[r] - xxq - xxb[m];
            if (negd > bestd[KNN - 1]) {
                // branch-free sorted insert (descending), constant indices only
#pragma unroll
                for (int s = KNN - 1; s > 0; --s) {
                    bool a1 = negd > bestd[s];
                    bool a0 = negd > bestd[s - 1];
                    float nd = a1 ? (a0 ? bestd[s - 1] : negd) : bestd[s];
                    int   ni = a1 ? (a0 ? besti[s - 1] : m)    : besti[s];
                    bestd[s] = nd; besti[s] = ni;
                }
                if (negd > bestd[0]) { bestd[0] = negd; besti[0] = m; }
            }
        }
    }

    __shared__ float sd[2][32][KNN];
    __shared__ int   si[2][32][KNN];
#pragma unroll
    for (int s = 0; s < KNN; ++s) { sd[wave][lane][s] = bestd[s]; si[wave][lane][s] = besti[s]; }
    __syncthreads();

    if (lane < 16) {
        int i = 0, j = 0;
        int* op = idx + ((size_t)b * N + q) * KNN;
#pragma unroll
        for (int t = 0; t < KNN; ++t) {
            float d0 = sd[wave][lane][i];
            float d1 = sd[wave][lane + 16][j];
            if (d0 >= d1) { op[t] = si[wave][lane][i]; ++i; }
            else          { op[t] = si[wave][lane + 16][j]; ++j; }
        }
    }
}

// ---------------------------------------------------------------------------
// Kernel 3: edge conv = max_j mish(bn(W @ [nbr-ctr;ctr])).
// CP = per-feature pad, MT = out_channels/16, edge K = 2*CP.
// Outputs yh (B,N,MT*16) f16 point-major + optional yxx (sum of squares).
// Block = 64 threads (2 waves, each its own 16-point tile).
// ---------------------------------------------------------------------------
template<int CP, int MT>
__global__ void edge_conv_kernel(const _Float16* __restrict__ xh, const int* __restrict__ idx,
                                 const _Float16* __restrict__ Wp,
                                 const float* __restrict__ gg, const float* __restrict__ bb,
                                 _Float16* __restrict__ yh, float* __restrict__ yxx, int N) {
    constexpr int KE = 2 * CP;
    int wave = threadIdx.x >> 5, lane = threadIdx.x & 31;
    int tile = blockIdx.x * 2 + wave;
    int tilesPerB = N / 16;
    int b = tile / tilesPerB, pt = tile - b * tilesPerB;

    const _Float16* xb = xh + (size_t)b * N * CP;
    int n = pt * 16 + (lane & 15);
    const int* nbrs = idx + ((size_t)b * N + n) * KNN;
    const _Float16* ctr = xb + (size_t)n * CP;

    float sq = 0.0f;
#pragma unroll
    for (int mt = 0; mt < MT; ++mt) {
        v16h a[KE / 32];
#pragma unroll
        for (int kc = 0; kc < KE / 32; ++kc)
            a[kc] = load_frag_a(Wp + (size_t)mt * 16 * KE + kc * 32, KE, lane);

        float mx[8];
#pragma unroll
        for (int r = 0; r < 8; ++r) mx[r] = -3.0e38f;

        for (int j = 0; j < KNN; ++j) {
            int nj = nbrs[j];
            if (j + 1 < KNN) __builtin_prefetch(xb + (size_t)nbrs[j + 1] * CP, 0, 1);
            const _Float16* nb = xb + (size_t)nj * CP;
            v8f acc = {};
#pragma unroll
            for (int kc = 0; kc < KE; kc += 32) {
                v16h bf = make_edge_bfrag<CP>(ctr, nb, kc, lane);
                acc = wmma_f16(a[kc / 32], bf, acc);
            }
#pragma unroll
            for (int r = 0; r < 8; ++r) {
                int m = mt * 16 + ((lane >> 4) << 3) + r;
                float y = acc[r] * (gg[m] * BN_RSQRT) + bb[m];
                mx[r] = fmaxf(mx[r], mish_f(y));
            }
        }
#pragma unroll
        for (int r = 0; r < 8; ++r) {
            int m = mt * 16 + ((lane >> 4) << 3) + r;
            yh[((size_t)b * N + n) * (MT * 16) + m] = (_Float16)mx[r];
            sq += mx[r] * mx[r];
        }
    }

    __shared__ float swp[2][32];
    swp[wave][lane] = sq;
    __syncthreads();
    if (lane < 16 && yxx != nullptr)
        yxx[(size_t)b * N + n] = swp[wave][lane] + swp[wave][lane + 16];
}

// ---------------------------------------------------------------------------
// Kernel 4: fused pointwise MLP chain (5 WMMA stages + f32 head), staged
// through LDS (stride 256 per point). Block = 64 threads (2 waves).
// ---------------------------------------------------------------------------
template<int KIN, int MOUT>
__device__ __forceinline__ void mlp_stage(const _Float16* __restrict__ Wp,
                                          const float* __restrict__ gg,
                                          const float* __restrict__ bb,
                                          const _Float16* in_lds, _Float16* out_lds,
                                          int lane) {
#pragma unroll
    for (int mt = 0; mt < MOUT / 16; ++mt) {
        v8f acc = {};
#pragma unroll
        for (int kc = 0; kc < KIN; kc += 32) {
            v16h a  = load_frag_a(Wp + (size_t)mt * 16 * KIN + kc, KIN, lane);
            v16h bf = load_frag_b(in_lds + kc, 256, lane);
            acc = wmma_f16(a, bf, acc);
        }
        int n = lane & 15;
#pragma unroll
        for (int r = 0; r < 8; ++r) {
            int m = mt * 16 + ((lane >> 4) << 3) + r;
            float y = acc[r] * (gg[m] * BN_RSQRT) + bb[m];
            out_lds[n * 256 + m] = (_Float16)mish_f(y);
        }
    }
}

__global__ void mlp_kernel(const _Float16* __restrict__ x1h, const _Float16* __restrict__ x2h,
                           const _Float16* __restrict__ x3h,
                           const _Float16* __restrict__ W4p, const _Float16* __restrict__ W5p,
                           const _Float16* __restrict__ W6p, const _Float16* __restrict__ W7p,
                           const _Float16* __restrict__ W8p, const float* __restrict__ W9,
                           const float* g4, const float* b4, const float* g5, const float* b5,
                           const float* g6, const float* b6, const float* g7, const float* b7,
                           const float* g8, const float* b8,
                           float* __restrict__ out, int N) {
    __shared__ __align__(16) _Float16 buf0[2][16 * 256];
    __shared__ __align__(16) _Float16 buf1[2][16 * 256];
    int wave = threadIdx.x >> 5, lane = threadIdx.x & 31;
    int tile = blockIdx.x * 2 + wave;
    int tilesPerB = N / 16;
    int b = tile / tilesPerB, pt = tile - b * tilesPerB;
    _Float16* B0 = buf0[wave];
    _Float16* B1 = buf1[wave];

    // load x3 (128 ch / point) via async memory->LDS when available
    const _Float16* x3p = x3h + ((size_t)b * N + pt * 16) * 128;
    stage_to_lds<128>(B0, x3p, 16, 256, lane);
    __syncthreads();

    mlp_stage<128, 128>(W4p, g4, b4, B0, B1, lane);           // h4 -> B1[0..127]
    __syncthreads();
    for (int t = lane; t < 16 * 128; t += 32) { int p = t >> 7, c = t & 127; B1[p * 256 + 128 + c] = B0[p * 256 + c]; }
    __syncthreads();

    mlp_stage<256, 128>(W5p, g5, b5, B1, B0, lane);           // -> B0[0..127]
    __syncthreads();
    mlp_stage<128, 64>(W6p, g6, b6, B0, B1, lane);            // -> B1[0..63]
    const _Float16* x2p = x2h + ((size_t)b * N + pt * 16) * 64;
    stage_to_lds<64>(B1 + 64, x2p, 16, 256, lane);            // cat x2 -> B1[64..127]
    __syncthreads();

    mlp_stage<128, 64>(W7p, g7, b7, B1, B0, lane);            // -> B0[0..63]
    const _Float16* x1p = x1h + ((size_t)b * N + pt * 16) * 64;
    stage_to_lds<64>(B0 + 64, x1p, 16, 256, lane);            // cat x1 -> B0[64..127]
    __syncthreads();

    mlp_stage<128, 64>(W8p, g8, b8, B0, B1, lane);            // -> B1[0..63]
    __syncthreads();

    // head: W9 (2x64), f32 scalar — 32 lanes = 16 points x 2 outputs
    {
        int p = lane & 15, o = lane >> 4;
        float s = 0.0f;
#pragma unroll
        for (int c = 0; c < 64; ++c) s += W9[o * 64 + c] * (float)B1[p * 256 + c];
        int n = pt * 16 + p;
        out[((size_t)b * 2 + o) * N + n] = s;
    }
}

// ---------------------------------------------------------------------------
// Launch
// ---------------------------------------------------------------------------
extern "C" void kernel_launch(void* const* d_in, const int* in_sizes, int n_in,
                              void* d_out, int out_size, void* d_ws, size_t ws_size,
                              hipStream_t stream) {
    const float* x  = (const float*)d_in[0];
    const float* W1 = (const float*)d_in[1];
    const float* W2 = (const float*)d_in[2];
    const float* W3 = (const float*)d_in[3];
    const float* W4 = (const float*)d_in[4];
    const float* W5 = (const float*)d_in[5];
    const float* W6 = (const float*)d_in[6];
    const float* W7 = (const float*)d_in[7];
    const float* W8 = (const float*)d_in[8];
    const float* W9 = (const float*)d_in[9];
    const float* g[8];
    const float* bn[8];
    for (int i = 0; i < 8; ++i) {
        g[i]  = (const float*)d_in[10 + 2 * i];
        bn[i] = (const float*)d_in[11 + 2 * i];
    }
    float* out = (float*)d_out;

    const int B = BATCH;
    const int N = in_sizes[0] / (3 * B);

    // workspace carve-out
    char* ws = (char*)d_ws;
    size_t off = 0;
    auto carve = [&](size_t bytes) -> void* {
        void* p = ws + off;
        off += (bytes + 255) & ~(size_t)255;
        return p;
    };
    _Float16* xh0 = (_Float16*)carve((size_t)B * N * 32 * 2);
    float*    xx0 = (float*)carve((size_t)B * N * 4);
    _Float16* x1h = (_Float16*)carve((size_t)B * N * 64 * 2);
    float*    xx1 = (float*)carve((size_t)B * N * 4);
    _Float16* x2h = (_Float16*)carve((size_t)B * N * 64 * 2);
    float*    xx2 = (float*)carve((size_t)B * N * 4);
    _Float16* x3h = (_Float16*)carve((size_t)B * N * 128 * 2);
    int*      idx = (int*)carve((size_t)B * N * KNN * 4);
    _Float16* W1p = (_Float16*)carve((size_t)64 * 64 * 2);
    _Float16* W2p = (_Float16*)carve((size_t)64 * 128 * 2);
    _Float16* W3p = (_Float16*)carve((size_t)128 * 128 * 2);
    _Float16* W4p = (_Float16*)carve((size_t)128 * 128 * 2);
    _Float16* W5p = (_Float16*)carve((size_t)128 * 256 * 2);
    _Float16* W6p = (_Float16*)carve((size_t)64 * 128 * 2);
    _Float16* W7p = (_Float16*)carve((size_t)64 * 128 * 2);
    _Float16* W8p = (_Float16*)carve((size_t)64 * 128 * 2);

    auto prep = [&](const float* W, _Float16* Wp, int O, int I, int IP, int split, int CPv) {
        int tot = O * IP;
        prep_weight_kernel<<<(tot + 255) / 256, 256, 0, stream>>>(W, Wp, O, I, IP, split, CPv);
    };
    prep(W1, W1p, 64, 6,   64,  3, 32);   // edge conv1: split diff/ctr to 32-pad halves
    prep(W2, W2p, 64, 128, 128, 0, 0);
    prep(W3, W3p, 128, 128, 128, 0, 0);
    prep(W4, W4p, 128, 128, 128, 0, 0);
    prep(W5, W5p, 128, 256, 256, 0, 0);
    prep(W6, W6p, 64, 128, 128, 0, 0);
    prep(W7, W7p, 64, 128, 128, 0, 0);
    prep(W8, W8p, 64, 128, 128, 0, 0);

    convert_x_kernel<<<(B * N + 255) / 256, 256, 0, stream>>>(x, xh0, xx0, N);

    int tileBlocks = (B * (N / 16)) / 2;   // 2 waves per 64-thread block

    // layer 1
    knn_kernel<32><<<tileBlocks, 64, 0, stream>>>(xh0, xx0, idx, N);
    edge_conv_kernel<32, 4><<<tileBlocks, 64, 0, stream>>>(xh0, idx, W1p, g[0], bn[0], x1h, xx1, N);
    // layer 2
    knn_kernel<64><<<tileBlocks, 64, 0, stream>>>(x1h, xx1, idx, N);
    edge_conv_kernel<64, 4><<<tileBlocks, 64, 0, stream>>>(x1h, idx, W2p, g[1], bn[1], x2h, xx2, N);
    // layer 3
    knn_kernel<64><<<tileBlocks, 64, 0, stream>>>(x2h, xx2, idx, N);
    edge_conv_kernel<64, 8><<<tileBlocks, 64, 0, stream>>>(x2h, idx, W3p, g[2], bn[2], x3h, nullptr, N);
    // fused MLP chain + head
    mlp_kernel<<<tileBlocks, 64, 0, stream>>>(x1h, x2h, x3h,
                                              W4p, W5p, W6p, W7p, W8p, W9,
                                              g[3], bn[3], g[4], bn[4], g[5], bn[5],
                                              g[6], bn[6], g[7], bn[7],
                                              out, N);
    (void)n_in; (void)out_size; (void)ws_size;
}
